// DeepImplicitEGNN_44796508897958
// MI455X (gfx1250) — matrix operand order repaired and checked by reading, not compiled, not run on hardware
//
#include <hip/hip_runtime.h>
#include <hip/hip_bf16.h>

typedef __attribute__((ext_vector_type(16))) _Float16 v16h;
typedef __attribute__((ext_vector_type(8)))  _Float16 v8h;
typedef __attribute__((ext_vector_type(4)))  _Float16 v4h;
typedef __attribute__((ext_vector_type(8)))  float    v8f;

#define WAVES 4      // waves per block
#define TM    32     // rows (edges/nodes) per wave = two 16-row M-tiles
#define EPS 1e-5f

// ---------------------------------------------------------------------------
// Helpers
// ---------------------------------------------------------------------------
__device__ __forceinline__ float silu_f(float x) {
    return x / (1.0f + __expf(-x));
}

// Build a WMMA A operand (16x32 f16, M=row striped) from an LDS tile stored
// row-major [rows][stride] f16.  Layout (ISA 7.12.2, 16-bit A 16x32):
//   lanes 0-15 : row = lane,     elems 0..7 = K kb..kb+7,  8..15 = K kb+16..kb+23
//   lanes16-31 : row = lane-16,  elems 0..7 = K kb+8..+15, 8..15 = K kb+24..+31
__device__ __forceinline__ v16h load_A(const _Float16* t, int lane, int kbase, int stride) {
    int row = lane & 15;
    int kh  = (lane < 16) ? 0 : 8;
    const _Float16* p = t + row * stride + kbase + kh;
    v8h lo = *(const v8h*)(p);
    v8h hi = *(const v8h*)(p + 16);
    v16h a;
#pragma unroll
    for (int i = 0; i < 8; ++i) { a[i] = lo[i]; a[i + 8] = hi[i]; }
    return a;
}

// B operands pre-packed: block = (kstep*ntiles + ntile); each lane owns 16
// contiguous f16 (32 B): elem i -> K = kstep*32 + (i<8?0:16) + (lane<16?0:8) + (i&7),
//                        N = ntile*16 + (lane&15)
__device__ __forceinline__ v16h load_B(const _Float16* packed, int block, int lane) {
    return *(const v16h*)(packed + ((size_t)block * 32 + lane) * 16);
}

__device__ __forceinline__ v8f wmma_f16(v16h a, v16h b, v8f c) {
    return __builtin_amdgcn_wmma_f32_16x16x32_f16(false, a, false, b, (short)0, c, false, false);
}

// ---------------------------------------------------------------------------
// One-time setup kernels
// ---------------------------------------------------------------------------
__global__ void pack_b_f16(const float* __restrict__ W, _Float16* __restrict__ dst,
                           int K, int N) {
    int t = blockIdx.x * blockDim.x + threadIdx.x;
    if (t >= K * N) return;
    int ntiles = N >> 4;
    int idx   = t & 15;
    int lane  = (t >> 4) & 31;
    int block = t >> 9;
    int ks = block / ntiles;
    int nt = block - ks * ntiles;
    int k = ks * 32 + ((idx < 8) ? 0 : 16) + ((lane < 16) ? 0 : 8) + (idx & 7);
    int n = nt * 16 + (lane & 15);
    dst[t] = (_Float16)W[k * N + n];
}

__global__ void d2_kernel(const float* __restrict__ pos, const int* __restrict__ ei,
                          float* __restrict__ d2g, int E) {
    int e = blockIdx.x * blockDim.x + threadIdx.x;
    if (e >= E) return;
    int s = ei[e];
    int r = ei[E + e];
    float acc = 0.f;
#pragma unroll
    for (int k = 0; k < 3; ++k) {
        float d = pos[s * 3 + k] - pos[r * 3 + k];
        acc += d * d;
    }
    d2g[e] = acc;
}

__global__ void zero_f32(float* __restrict__ p, int n) {
    int i = blockIdx.x * blockDim.x + threadIdx.x;
    if (i < n) p[i] = 0.f;
}

// ---------------------------------------------------------------------------
// Edge kernel: fused 3-layer edge MLP + atomic scatter into agg.
// One wave = 32 edges (2 M-tiles, shared B operands). 4 waves/block.
// All LDS is wave-private -> no barriers; EXEC stays full for WMMA.
// ---------------------------------------------------------------------------
__global__ __launch_bounds__(128) void edge_kernel(
    const float* __restrict__ z, const int* __restrict__ ei,
    const float* __restrict__ d2g,
    const _Float16* __restrict__ Wp1, const float* __restrict__ b_e1,
    const float* __restrict__ w128,
    const _Float16* __restrict__ Wp2, const float* __restrict__ b_e2,
    const _Float16* __restrict__ Wp3, const float* __restrict__ b_e3,
    float* __restrict__ agg, int E)
{
    __shared__ _Float16 ein[WAVES][TM][128];
    __shared__ _Float16 h1 [WAVES][TM][64];
    __shared__ _Float16 h2 [WAVES][TM][32];
    __shared__ float    d2s[WAVES][TM];
    __shared__ int      rs [WAVES][TM];

    int lane = threadIdx.x & 31;
    int w    = threadIdx.x >> 5;
    int ebase = blockIdx.x * (WAVES * TM) + w * TM;
    if (ebase >= E) return;                 // uniform per wave (E % TM == 0)

    d2s[w][lane] = d2g[ebase + lane];
    rs [w][lane] = ei[E + ebase + lane];

    // Gather [z_r | z_s] -> f16 LDS tile [TM][128].
    // Lanes 0-15 fetch a float4 of z[r] (cols 0..63); lanes 16-31 of z[s].
    for (int i = 0; i < TM; ++i) {
        int e  = ebase + i;
        int sN = ei[e];
        int rN = ei[E + e];
        int src = (lane < 16) ? rN : sN;
        float4 vv = ((const float4*)(z + (size_t)src * 64))[lane & 15];
        v4h hh;
        hh[0] = (_Float16)vv.x; hh[1] = (_Float16)vv.y;
        hh[2] = (_Float16)vv.z; hh[3] = (_Float16)vv.w;
        *(v4h*)(&ein[w][i][lane * 4]) = hh;
    }

    int col  = lane & 15;
    int roff = (lane < 16) ? 0 : 8;

    // ---- GEMM1: [32,128] x [128,64], C0 = b_e1 + d2 * W_e1[128,:] ----
    v8f c1[2][4];
#pragma unroll
    for (int nt = 0; nt < 4; ++nt) {
        float bb = b_e1[nt * 16 + col];
        float ww = w128[nt * 16 + col];
#pragma unroll
        for (int m = 0; m < 2; ++m)
#pragma unroll
            for (int j = 0; j < 8; ++j)
                c1[m][nt][j] = bb + d2s[w][m * 16 + roff + j] * ww;
    }
#pragma unroll
    for (int ks = 0; ks < 4; ++ks) {
        v16h a0 = load_A(&ein[w][0][0],  lane, ks * 32, 128);
        v16h a1 = load_A(&ein[w][16][0], lane, ks * 32, 128);
#pragma unroll
        for (int nt = 0; nt < 4; ++nt) {
            v16h b = load_B(Wp1, ks * 4 + nt, lane);
            c1[0][nt] = wmma_f16(a0, b, c1[0][nt]);
            c1[1][nt] = wmma_f16(a1, b, c1[1][nt]);
        }
    }
#pragma unroll
    for (int m = 0; m < 2; ++m)
#pragma unroll
        for (int nt = 0; nt < 4; ++nt)
#pragma unroll
            for (int j = 0; j < 8; ++j)
                h1[w][m * 16 + roff + j][nt * 16 + col] = (_Float16)silu_f(c1[m][nt][j]);

    // ---- GEMM2: [32,64] x [64,32] ----
    v8f c2[2][2];
#pragma unroll
    for (int nt = 0; nt < 2; ++nt) {
        float bb = b_e2[nt * 16 + col];
#pragma unroll
        for (int m = 0; m < 2; ++m)
#pragma unroll
            for (int j = 0; j < 8; ++j) c2[m][nt][j] = bb;
    }
#pragma unroll
    for (int ks = 0; ks < 2; ++ks) {
        v16h a0 = load_A(&h1[w][0][0],  lane, ks * 32, 64);
        v16h a1 = load_A(&h1[w][16][0], lane, ks * 32, 64);
#pragma unroll
        for (int nt = 0; nt < 2; ++nt) {
            v16h b = load_B(Wp2, ks * 2 + nt, lane);
            c2[0][nt] = wmma_f16(a0, b, c2[0][nt]);
            c2[1][nt] = wmma_f16(a1, b, c2[1][nt]);
        }
    }
#pragma unroll
    for (int m = 0; m < 2; ++m)
#pragma unroll
        for (int nt = 0; nt < 2; ++nt)
#pragma unroll
            for (int j = 0; j < 8; ++j)
                h2[w][m * 16 + roff + j][nt * 16 + col] = (_Float16)silu_f(c2[m][nt][j]);

    // ---- GEMM3: [32,32] x [32,64], atomic scatter into agg[r] ----
    v8f c3[2][4];
#pragma unroll
    for (int nt = 0; nt < 4; ++nt) {
        float bb = b_e3[nt * 16 + col];
#pragma unroll
        for (int m = 0; m < 2; ++m)
#pragma unroll
            for (int j = 0; j < 8; ++j) c3[m][nt][j] = bb;
    }
    {
        v16h a0 = load_A(&h2[w][0][0],  lane, 0, 32);
        v16h a1 = load_A(&h2[w][16][0], lane, 0, 32);
#pragma unroll
        for (int nt = 0; nt < 4; ++nt) {
            v16h b = load_B(Wp3, nt, lane);
            c3[0][nt] = wmma_f16(a0, b, c3[0][nt]);
            c3[1][nt] = wmma_f16(a1, b, c3[1][nt]);
        }
    }
#pragma unroll
    for (int m = 0; m < 2; ++m)
#pragma unroll
        for (int nt = 0; nt < 4; ++nt)
#pragma unroll
            for (int j = 0; j < 8; ++j) {
                int r = rs[w][m * 16 + roff + j];
                atomicAdd(&agg[r * 64 + nt * 16 + col], c3[m][nt][j]);
            }
}

// ---------------------------------------------------------------------------
// Node kernel: fused node MLP + BN0 + x-injection + BN1 -> z_next
// One wave = 32 nodes (2 M-tiles). 4 waves/block.
// ---------------------------------------------------------------------------
__global__ __launch_bounds__(128) void node_kernel(
    const float* __restrict__ z, const float* __restrict__ agg,
    const float* __restrict__ xin,
    const _Float16* __restrict__ Wph1, const float* __restrict__ b_h1,
    const _Float16* __restrict__ Wph2, const float* __restrict__ b_h2,
    const float* __restrict__ g0, const float* __restrict__ b0,
    const float* __restrict__ m0, const float* __restrict__ v0,
    const float* __restrict__ g1, const float* __restrict__ b1,
    const float* __restrict__ m1, const float* __restrict__ v1,
    float* __restrict__ zout, int Nn)
{
    __shared__ _Float16 uin[WAVES][TM][128];
    __shared__ _Float16 u  [WAVES][TM][32];

    int lane = threadIdx.x & 31;
    int w    = threadIdx.x >> 5;
    int nbase = blockIdx.x * (WAVES * TM) + w * TM;
    if (nbase >= Nn) return;                // uniform per wave (N % TM == 0)

    // Lanes 0-15 fetch float4 of z[n] (cols 0..63); lanes 16-31 of agg[n].
    for (int i = 0; i < TM; ++i) {
        int n = nbase + i;
        const float* srcp = (lane < 16) ? (z + (size_t)n * 64) : (agg + (size_t)n * 64);
        float4 vv = ((const float4*)srcp)[lane & 15];
        v4h hh;
        hh[0] = (_Float16)vv.x; hh[1] = (_Float16)vv.y;
        hh[2] = (_Float16)vv.z; hh[3] = (_Float16)vv.w;
        *(v4h*)(&uin[w][i][lane * 4]) = hh;
    }

    int col  = lane & 15;
    int roff = (lane < 16) ? 0 : 8;

    // ---- GEMM: [32,128] x [128,32] ----
    v8f c1[2][2];
#pragma unroll
    for (int nt = 0; nt < 2; ++nt) {
        float bb = b_h1[nt * 16 + col];
#pragma unroll
        for (int m = 0; m < 2; ++m)
#pragma unroll
            for (int j = 0; j < 8; ++j) c1[m][nt][j] = bb;
    }
#pragma unroll
    for (int ks = 0; ks < 4; ++ks) {
        v16h a0 = load_A(&uin[w][0][0],  lane, ks * 32, 128);
        v16h a1 = load_A(&uin[w][16][0], lane, ks * 32, 128);
#pragma unroll
        for (int nt = 0; nt < 2; ++nt) {
            v16h b = load_B(Wph1, ks * 2 + nt, lane);
            c1[0][nt] = wmma_f16(a0, b, c1[0][nt]);
            c1[1][nt] = wmma_f16(a1, b, c1[1][nt]);
        }
    }
#pragma unroll
    for (int m = 0; m < 2; ++m)
#pragma unroll
        for (int nt = 0; nt < 2; ++nt)
#pragma unroll
            for (int j = 0; j < 8; ++j)
                u[w][m * 16 + roff + j][nt * 16 + col] = (_Float16)silu_f(c1[m][nt][j]);

    // ---- GEMM: [32,32] x [32,64] ----
    v8f h[2][4];
#pragma unroll
    for (int nt = 0; nt < 4; ++nt) {
        float bb = b_h2[nt * 16 + col];
#pragma unroll
        for (int m = 0; m < 2; ++m)
#pragma unroll
            for (int j = 0; j < 8; ++j) h[m][nt][j] = bb;
    }
    {
        v16h a0 = load_A(&u[w][0][0],  lane, 0, 32);
        v16h a1 = load_A(&u[w][16][0], lane, 0, 32);
#pragma unroll
        for (int nt = 0; nt < 4; ++nt) {
            v16h b = load_B(Wph2, nt, lane);
            h[0][nt] = wmma_f16(a0, b, h[0][nt]);
            h[1][nt] = wmma_f16(a1, b, h[1][nt]);
        }
    }

    // ---- BN0 -> +x -> BN1 -> store ----
#pragma unroll
    for (int nt = 0; nt < 4; ++nt) {
        int cc = nt * 16 + col;
        float s0 = rsqrtf(v0[cc] + EPS) * g0[cc];
        float a0 = b0[cc] - m0[cc] * s0;
        float s1 = rsqrtf(v1[cc] + EPS) * g1[cc];
        float a1 = b1[cc] - m1[cc] * s1;
#pragma unroll
        for (int m = 0; m < 2; ++m)
#pragma unroll
            for (int j = 0; j < 8; ++j) {
                int n = nbase + m * 16 + roff + j;
                float hv = h[m][nt][j];
                hv = hv * s0 + a0;
                hv = hv + xin[n * 64 + cc];
                hv = hv * s1 + a1;
                zout[n * 64 + cc] = hv;
            }
    }
}

// ---------------------------------------------------------------------------
// Host-side launch
// ---------------------------------------------------------------------------
extern "C" void kernel_launch(void* const* d_in, const int* in_sizes, int n_in,
                              void* d_out, int out_size, void* d_ws, size_t ws_size,
                              hipStream_t stream) {
    const float* x    = (const float*)d_in[0];
    const float* pos  = (const float*)d_in[1];
    const int*   ei   = (const int*)d_in[2];
    const float* W_e1 = (const float*)d_in[3];
    const float* b_e1 = (const float*)d_in[4];
    const float* W_e2 = (const float*)d_in[5];
    const float* b_e2 = (const float*)d_in[6];
    const float* W_e3 = (const float*)d_in[7];
    const float* b_e3 = (const float*)d_in[8];
    const float* W_h1 = (const float*)d_in[9];
    const float* b_h1 = (const float*)d_in[10];
    const float* W_h2 = (const float*)d_in[11];
    const float* b_h2 = (const float*)d_in[12];
    const float* g0 = (const float*)d_in[13];
    const float* b0 = (const float*)d_in[14];
    const float* m0 = (const float*)d_in[15];
    const float* v0 = (const float*)d_in[16];
    const float* g1 = (const float*)d_in[17];
    const float* b1 = (const float*)d_in[18];
    const float* m1 = (const float*)d_in[19];
    const float* v1 = (const float*)d_in[20];

    const int Nn = in_sizes[0] / 64;     // 20000
    const int E  = in_sizes[2] / 2;      // 320000
    const int F  = 64;
    const int ITERS = 30;

    // --- workspace layout (bytes) ---
    char* ws = (char*)d_ws;
    _Float16* Wp1  = (_Float16*)(ws + 0);          // 128x64  -> 16384 B
    _Float16* Wp2  = (_Float16*)(ws + 16384);      // 64x32   ->  4096 B
    _Float16* Wp3  = (_Float16*)(ws + 20480);      // 32x64   ->  4096 B
    _Float16* Wph1 = (_Float16*)(ws + 24576);      // 128x32  ->  8192 B
    _Float16* Wph2 = (_Float16*)(ws + 32768);      // 32x64   ->  4096 B
    size_t off = 36864;
    float* d2g = (float*)(ws + off);  off += (size_t)E * 4;
    float* agg = (float*)(ws + off);  off += (size_t)Nn * F * 4;
    float* zA  = (float*)(ws + off);  off += (size_t)Nn * F * 4;
    float* zB  = (float*)(ws + off);  off += (size_t)Nn * F * 4;
    (void)ws_size;

    const float* w128 = W_e1 + 128 * 64;  // d2 row of W_e1 (row 128 of [129,64])

    // --- one-time setup ---
    pack_b_f16<<<(128 * 64 + 255) / 256, 256, 0, stream>>>(W_e1, Wp1, 128, 64);
    pack_b_f16<<<( 64 * 32 + 255) / 256, 256, 0, stream>>>(W_e2, Wp2,  64, 32);
    pack_b_f16<<<( 32 * 64 + 255) / 256, 256, 0, stream>>>(W_e3, Wp3,  32, 64);
    pack_b_f16<<<(128 * 32 + 255) / 256, 256, 0, stream>>>(W_h1, Wph1, 128, 32);
    pack_b_f16<<<( 32 * 64 + 255) / 256, 256, 0, stream>>>(W_h2, Wph2,  32, 64);
    d2_kernel<<<(E + 255) / 256, 256, 0, stream>>>(pos, ei, d2g, E);

    int nzf = Nn * F;
    zero_f32<<<(nzf + 255) / 256, 256, 0, stream>>>(zA, nzf);   // z0 = 0

    float* zcur = zA;
    float* znxt = zB;
    int rows_per_block = WAVES * TM;     // 128
    int edge_blocks = (E + rows_per_block - 1) / rows_per_block;
    int node_blocks = (Nn + rows_per_block - 1) / rows_per_block;

    for (int it = 0; it < ITERS; ++it) {
        zero_f32<<<(nzf + 255) / 256, 256, 0, stream>>>(agg, nzf);
        edge_kernel<<<edge_blocks, 128, 0, stream>>>(
            zcur, ei, d2g, Wp1, b_e1, w128, Wp2, b_e2, Wp3, b_e3, agg, E);
        float* out = (it == ITERS - 1) ? (float*)d_out : znxt;
        node_kernel<<<node_blocks, 128, 0, stream>>>(
            zcur, agg, x, Wph1, b_h1, Wph2, b_h2,
            g0, b0, m0, v0, g1, b1, m1, v1, out, Nn);
        float* t = zcur; zcur = znxt; znxt = t;
    }
}
